// EdgeConv_77171972375205
// MI455X (gfx1250) — compile-verified, matrix operand order checked by reading, not run on hardware
//
#include <hip/hip_runtime.h>

// ---------------------------------------------------------------------------
// EdgeConv (DGCNN) for MI455X / gfx1250.  B=2, N=4096, K=20, H=384.
// All large GEMMs via v_wmma_f32_16x16x32_bf16 (wave32), bf16 activations,
// BN statistics fused into producer passes, final max fused via u32 atomicMax.
// v3: dual accumulator chains (2 m-tiles in flight) + 1-stage software
// pipeline on A-fragment loads so s_wait_loadcnt is issued with work in
// flight instead of just-in-time (loads for k+1 overlap WMMAs for k).
// ---------------------------------------------------------------------------

typedef __bf16 bf16_t;
typedef __attribute__((ext_vector_type(16))) __bf16 v16bf;
typedef __attribute__((ext_vector_type(8)))  __bf16 v8bf;
typedef __attribute__((ext_vector_type(8)))  float  v8f;

#define BATCH   2
#define NPTS    4096
#define KNN     20
#define HDIM    384
#define MROWS   163840      // BATCH*NPTS*KNN
#define PPTS    8192        // BATCH*NPTS
#define BN_EPS  1e-5f

// ----- 16x32 bf16 fragment load (A layout; B uses same layout on W^T) -------
__device__ __forceinline__ v16bf frag_from(const bf16_t* p) {
  v8bf lo = *(const v8bf*)(p);
  v8bf hi = *(const v8bf*)(p + 16);
  v16bf r;
#pragma unroll
  for (int i = 0; i < 8; ++i) { r[i] = lo[i]; r[i + 8] = hi[i]; }
  return r;
}

__device__ __forceinline__ unsigned enc_f32(float f) {
  unsigned b = __float_as_uint(f);
  return (b & 0x80000000u) ? ~b : (b | 0x80000000u);
}

// ---------------------------------------------------------------------------
// 0) init: copy xyz -> out, set encoded -inf in feature region, zero BN stats
// ---------------------------------------------------------------------------
__global__ void init_kernel(const float* __restrict__ xyz, float* __restrict__ out_xyz,
                            unsigned* __restrict__ enc, float* __restrict__ stats1,
                            float* __restrict__ stats3) {
  size_t i = (size_t)blockIdx.x * blockDim.x + threadIdx.x;
  if (i < (size_t)BATCH * NPTS * 3) out_xyz[i] = xyz[i];
  if (i < (size_t)PPTS * HDIM)      enc[i] = 0x007FFFFFu;   // enc(-inf)
  if (i < 256)                      stats1[i] = 0.0f;
  if (i < 1024)                     stats3[i] = 0.0f;
}

// ---------------------------------------------------------------------------
// 1) weight transpose + fp32 -> bf16  (Wt[n*K+k] = W[k*N+n])
// ---------------------------------------------------------------------------
__global__ void transpose_bf16_kernel(const float* __restrict__ W, bf16_t* __restrict__ Wt,
                                      int Kdim, int Ndim) {
  int t = blockIdx.x * blockDim.x + threadIdx.x;
  if (t >= Kdim * Ndim) return;
  int n = t / Kdim, k = t - n * Kdim;
  Wt[t] = (bf16_t)W[(size_t)k * Ndim + n];
}

// ---------------------------------------------------------------------------
// 2) kNN: all points of one batch in LDS, register-resident top-20 insertion
// ---------------------------------------------------------------------------
__global__ void knn_kernel(const float* __restrict__ xyz, int* __restrict__ idx) {
  __shared__ float sx[NPTS], sy[NPTS], sz[NPTS];
  const int b = blockIdx.y;
  const float* base = xyz + (size_t)b * NPTS * 3;
  for (int t = threadIdx.x; t < NPTS; t += blockDim.x) {
    sx[t] = base[t * 3 + 0];
    sy[t] = base[t * 3 + 1];
    sz[t] = base[t * 3 + 2];
  }
  __syncthreads();
  const int i = blockIdx.x * blockDim.x + threadIdx.x;
  const float xi = sx[i], yi = sy[i], zi = sz[i];
  float bd[KNN]; int bi[KNN];
#pragma unroll
  for (int s = 0; s < KNN; ++s) { bd[s] = 3.0e38f; bi[s] = i; }
  for (int j = 0; j < NPTS; ++j) {
    float dx = sx[j] - xi, dy = sy[j] - yi, dz = sz[j] - zi;
    float d2 = dx * dx + dy * dy + dz * dz;
    if (d2 < bd[KNN - 1]) {
      float cd = d2; int ci = j;
#pragma unroll
      for (int s = 0; s < KNN; ++s) {
        if (cd < bd[s]) { float td = bd[s]; int ti = bi[s]; bd[s] = cd; bi[s] = ci; cd = td; ci = ti; }
      }
    }
  }
  int* op = idx + ((size_t)(b * NPTS + i)) * KNN;
#pragma unroll
  for (int s = 0; s < KNN; ++s) op[s] = bi[s];
}

// ---------------------------------------------------------------------------
// 3) edge features + MLP1 (6->128) + fused BN1 sum/sumsq;  h1 stored bf16
// ---------------------------------------------------------------------------
__global__ void __launch_bounds__(128) mlp1_kernel(const float* __restrict__ xyz,
                                                   const int* __restrict__ idx,
                                                   const float* __restrict__ W1,
                                                   const float* __restrict__ b1,
                                                   bf16_t* __restrict__ h1,
                                                   float* __restrict__ stats1) {
  __shared__ float edge[128][6];
  __shared__ float w1s[6 * 128];
  const int tid = threadIdx.x;
  const size_t r0 = (size_t)blockIdx.x * 128;
  for (int t = tid; t < 6 * 128; t += 128) w1s[t] = W1[t];
  {
    size_t r = r0 + tid;
    int p = (int)(r / KNN);
    int j = idx[r];
    int b = p >> 12;            // / NPTS
    int i = p & (NPTS - 1);
    const float* c  = xyz + ((size_t)(b * NPTS + i)) * 3;
    const float* nb = xyz + ((size_t)(b * NPTS + j)) * 3;
    edge[tid][0] = nb[0] - c[0];
    edge[tid][1] = nb[1] - c[1];
    edge[tid][2] = nb[2] - c[2];
    edge[tid][3] = c[0];
    edge[tid][4] = c[1];
    edge[tid][5] = c[2];
  }
  __syncthreads();
  const int ch = tid;
  const float w0 = w1s[0 * 128 + ch], w1 = w1s[1 * 128 + ch], w2 = w1s[2 * 128 + ch];
  const float w3 = w1s[3 * 128 + ch], w4 = w1s[4 * 128 + ch], w5 = w1s[5 * 128 + ch];
  const float bia = b1[ch];
  float s = 0.0f, sq = 0.0f;
  for (int rr = 0; rr < 128; ++rr) {
    float h = bia + edge[rr][0] * w0 + edge[rr][1] * w1 + edge[rr][2] * w2
                  + edge[rr][3] * w3 + edge[rr][4] * w4 + edge[rr][5] * w5;
    h1[(r0 + rr) * 128 + ch] = (bf16_t)h;
    s += h; sq += h * h;
  }
  atomicAdd(&stats1[ch], s);
  atomicAdd(&stats1[128 + ch], sq);
}

// ---------------------------------------------------------------------------
// 4) BN finalize + in-place bf16 BN+ReLU apply
// ---------------------------------------------------------------------------
__global__ void bn_finalize_kernel(const float* __restrict__ stats, int C, float invM,
                                   const float* __restrict__ g, const float* __restrict__ be,
                                   float* __restrict__ scale, float* __restrict__ shift) {
  int c = threadIdx.x;
  if (c >= C) return;
  float mu  = stats[c] * invM;
  float var = stats[C + c] * invM - mu * mu;
  float sc  = g[c] * rsqrtf(var + BN_EPS);
  scale[c] = sc;
  shift[c] = be[c] - mu * sc;
}

__global__ void bn_apply_kernel(bf16_t* __restrict__ buf, const float* __restrict__ scale,
                                const float* __restrict__ shift, unsigned cmask, size_t total) {
  size_t stride = (size_t)gridDim.x * blockDim.x;
  for (size_t i = (size_t)blockIdx.x * blockDim.x + threadIdx.x; i < total; i += stride) {
    unsigned c = (unsigned)i & cmask;
    float x = (float)buf[i];
    x = fmaxf(x * scale[c] + shift[c], 0.0f);
    buf[i] = (bf16_t)x;
  }
}

// ---------------------------------------------------------------------------
// 5) GEMM2: h2[M,256] = relu_bn(h1)[M,128] @ W2 + b2   (bf16 WMMA, f32 acc)
//    4 waves/block, 1 n-tile per wave, 2 m-tiles in flight, k-pipelined loads
// ---------------------------------------------------------------------------
__global__ void __launch_bounds__(128) gemm2_kernel(const bf16_t* __restrict__ A,
                                                    const bf16_t* __restrict__ Bt,
                                                    const float* __restrict__ bias,
                                                    bf16_t* __restrict__ D) {
  const int KDIM = 128, NDIM = 256, KT = 4, MT = 32;
  const int lane = threadIdx.x & 31;
  const int wave = threadIdx.x >> 5;
  const int n0 = (blockIdx.y * 4 + wave) * 16;
  const int mlane = lane & 15;
  const int koff = (lane >> 4) * 8;
  const int n = n0 + mlane;

  v16bf bfrag[KT];
#pragma unroll
  for (int kt = 0; kt < KT; ++kt)
    bfrag[kt] = frag_from(Bt + (size_t)n * KDIM + kt * 32 + koff);
  const float bval = bias[n];

  for (int mt = 0; mt < MT; mt += 2) {
    const int m0 = (blockIdx.x * MT + mt) * 16;
    const bf16_t* arow0 = A + (size_t)(m0 + mlane) * KDIM + koff;
    const bf16_t* arow1 = arow0 + (size_t)16 * KDIM;
    if (mt + 2 < MT) __builtin_prefetch(arow0 + (size_t)32 * KDIM, 0, 1);
    v8f acc0, acc1;
#pragma unroll
    for (int v = 0; v < 8; ++v) { acc0[v] = bval; acc1[v] = bval; }
    // 1-stage software pipeline over k-tiles
    v16bf a0 = frag_from(arow0);
    v16bf a1 = frag_from(arow1);
#pragma unroll
    for (int kt = 0; kt < KT; ++kt) {
      v16bf nx0 = a0, nx1 = a1;
      if (kt + 1 < KT) {
        nx0 = frag_from(arow0 + (kt + 1) * 32);
        nx1 = frag_from(arow1 + (kt + 1) * 32);
      }
      acc0 = __builtin_amdgcn_wmma_f32_16x16x32_bf16(false, a0, false, bfrag[kt],
                                                     (short)0, acc0, false, false);
      acc1 = __builtin_amdgcn_wmma_f32_16x16x32_bf16(false, a1, false, bfrag[kt],
                                                     (short)0, acc1, false, false);
      a0 = nx0; a1 = nx1;
    }
    const int mbase0 = m0 + ((lane >> 4) << 3);
#pragma unroll
    for (int v = 0; v < 8; ++v)
      D[(size_t)(mbase0 + v) * NDIM + n] = (bf16_t)acc0[v];
#pragma unroll
    for (int v = 0; v < 8; ++v)
      D[(size_t)(mbase0 + 16 + v) * NDIM + n] = (bf16_t)acc1[v];
  }
}

// ---------------------------------------------------------------------------
// 6) per-point max over K neighbors: gmax[P,256]
// ---------------------------------------------------------------------------
__global__ void gmax_kernel(const bf16_t* __restrict__ h2, bf16_t* __restrict__ gmax) {
  const int p = blockIdx.x, c = threadIdx.x;
  const bf16_t* base = h2 + (size_t)p * KNN * 256 + c;
  float m = (float)base[0];
#pragma unroll
  for (int k = 1; k < KNN; ++k) m = fmaxf(m, (float)base[(size_t)k * 256]);
  gmax[(size_t)p * 256 + c] = (bf16_t)m;
}

// ---------------------------------------------------------------------------
// 7) GEMM3: f3pre[M,512] = [gmax(p)||h2(r)][M,512] @ W3 + b3; fused BN3 stats
// ---------------------------------------------------------------------------
__device__ __forceinline__ const bf16_t* a3_ptr(const bf16_t* __restrict__ gmax,
                                                const bf16_t* __restrict__ h2,
                                                int kt, int r, int p, int koff) {
  const int k0 = kt * 32;
  return (k0 < 256) ? (gmax + (size_t)p * 256 + k0 + koff)
                    : (h2   + (size_t)r * 256 + (k0 - 256) + koff);
}

__global__ void __launch_bounds__(128) gemm3_kernel(const bf16_t* __restrict__ gmax,
                                                    const bf16_t* __restrict__ h2,
                                                    const bf16_t* __restrict__ Bt,
                                                    const float* __restrict__ bias,
                                                    bf16_t* __restrict__ D,
                                                    float* __restrict__ stats3) {
  const int KDIM = 512, NDIM = 512, KT = 16, MT = 32;
  const int lane = threadIdx.x & 31;
  const int wave = threadIdx.x >> 5;
  const int n0 = (blockIdx.y * 4 + wave) * 16;
  const int mlane = lane & 15;
  const int koff = (lane >> 4) * 8;
  const int n = n0 + mlane;

  v16bf bfrag[KT];
#pragma unroll
  for (int kt = 0; kt < KT; ++kt)
    bfrag[kt] = frag_from(Bt + (size_t)n * KDIM + kt * 32 + koff);
  const float bval = bias[n];

  float lsum = 0.0f, lsq = 0.0f;
  for (int mt = 0; mt < MT; mt += 2) {
    const int m0 = (blockIdx.x * MT + mt) * 16;
    const int r0 = m0 + mlane;
    const int r1 = r0 + 16;
    const int p0 = r0 / KNN;
    const int p1 = r1 / KNN;
    if (mt + 2 < MT) __builtin_prefetch(h2 + (size_t)(r0 + 32) * 256 + koff, 0, 1);
    v8f acc0, acc1;
#pragma unroll
    for (int v = 0; v < 8; ++v) { acc0[v] = bval; acc1[v] = bval; }
    v16bf a0 = frag_from(a3_ptr(gmax, h2, 0, r0, p0, koff));
    v16bf a1 = frag_from(a3_ptr(gmax, h2, 0, r1, p1, koff));
#pragma unroll
    for (int kt = 0; kt < KT; ++kt) {
      v16bf nx0 = a0, nx1 = a1;
      if (kt + 1 < KT) {
        nx0 = frag_from(a3_ptr(gmax, h2, kt + 1, r0, p0, koff));
        nx1 = frag_from(a3_ptr(gmax, h2, kt + 1, r1, p1, koff));
      }
      acc0 = __builtin_amdgcn_wmma_f32_16x16x32_bf16(false, a0, false, bfrag[kt],
                                                     (short)0, acc0, false, false);
      acc1 = __builtin_amdgcn_wmma_f32_16x16x32_bf16(false, a1, false, bfrag[kt],
                                                     (short)0, acc1, false, false);
      a0 = nx0; a1 = nx1;
    }
    const int mbase0 = m0 + ((lane >> 4) << 3);
#pragma unroll
    for (int v = 0; v < 8; ++v) {
      float x = acc0[v];
      D[(size_t)(mbase0 + v) * NDIM + n] = (bf16_t)x;
      lsum += x; lsq += x * x;
    }
#pragma unroll
    for (int v = 0; v < 8; ++v) {
      float x = acc1[v];
      D[(size_t)(mbase0 + 16 + v) * NDIM + n] = (bf16_t)x;
      lsum += x; lsq += x * x;
    }
  }
  atomicAdd(&stats3[n], lsum);
  atomicAdd(&stats3[512 + n], lsq);
}

// ---------------------------------------------------------------------------
// 8) GEMM4: [M,384] = relu_bn(f3pre)[M,512] @ W4 + b4, fused max over K
//    via monotone-encoded u32 atomicMax (global_atomic_max_u32).
// ---------------------------------------------------------------------------
__device__ __forceinline__ void maxk_flush(const v8f& acc, int mbase, int n,
                                           unsigned* __restrict__ enc_out) {
  float cur = acc[0];
  int curp = mbase / KNN;
#pragma unroll
  for (int v = 1; v < 8; ++v) {
    int p = (mbase + v) / KNN;
    if (p != curp) {
      atomicMax(&enc_out[(size_t)curp * HDIM + n], enc_f32(cur));
      curp = p; cur = acc[v];
    } else {
      cur = fmaxf(cur, acc[v]);
    }
  }
  atomicMax(&enc_out[(size_t)curp * HDIM + n], enc_f32(cur));
}

__global__ void __launch_bounds__(128) gemm4_kernel(const bf16_t* __restrict__ A,
                                                    const bf16_t* __restrict__ Bt,
                                                    const float* __restrict__ bias,
                                                    unsigned* __restrict__ enc_out) {
  const int KDIM = 512, KT = 16, MT = 32;
  const int lane = threadIdx.x & 31;
  const int wave = threadIdx.x >> 5;
  const int n0 = (blockIdx.y * 4 + wave) * 16;
  const int mlane = lane & 15;
  const int koff = (lane >> 4) * 8;
  const int n = n0 + mlane;

  v16bf bfrag[KT];
#pragma unroll
  for (int kt = 0; kt < KT; ++kt)
    bfrag[kt] = frag_from(Bt + (size_t)n * KDIM + kt * 32 + koff);
  const float bval = bias[n];

  for (int mt = 0; mt < MT; mt += 2) {
    const int m0 = (blockIdx.x * MT + mt) * 16;
    const bf16_t* arow0 = A + (size_t)(m0 + mlane) * KDIM + koff;
    const bf16_t* arow1 = arow0 + (size_t)16 * KDIM;
    if (mt + 2 < MT) __builtin_prefetch(arow0 + (size_t)32 * KDIM, 0, 1);
    v8f acc0, acc1;
#pragma unroll
    for (int v = 0; v < 8; ++v) { acc0[v] = bval; acc1[v] = bval; }
    v16bf a0 = frag_from(arow0);
    v16bf a1 = frag_from(arow1);
#pragma unroll
    for (int kt = 0; kt < KT; ++kt) {
      v16bf nx0 = a0, nx1 = a1;
      if (kt + 1 < KT) {
        nx0 = frag_from(arow0 + (kt + 1) * 32);
        nx1 = frag_from(arow1 + (kt + 1) * 32);
      }
      acc0 = __builtin_amdgcn_wmma_f32_16x16x32_bf16(false, a0, false, bfrag[kt],
                                                     (short)0, acc0, false, false);
      acc1 = __builtin_amdgcn_wmma_f32_16x16x32_bf16(false, a1, false, bfrag[kt],
                                                     (short)0, acc1, false, false);
      a0 = nx0; a1 = nx1;
    }
    const int mbase0 = m0 + ((lane >> 4) << 3);
    maxk_flush(acc0, mbase0, n, enc_out);
    maxk_flush(acc1, mbase0 + 16, n, enc_out);
  }
}

// ---------------------------------------------------------------------------
// 9) decode encoded-u32 maxima back to fp32 in place
// ---------------------------------------------------------------------------
__global__ void decode_kernel(unsigned* __restrict__ enc, size_t total) {
  size_t i = (size_t)blockIdx.x * blockDim.x + threadIdx.x;
  if (i >= total) return;
  unsigned u = enc[i];
  unsigned b = (u & 0x80000000u) ? (u & 0x7FFFFFFFu) : ~u;
  ((float*)enc)[i] = __uint_as_float(b);
}

// ---------------------------------------------------------------------------
extern "C" void kernel_launch(void* const* d_in, const int* in_sizes, int n_in,
                              void* d_out, int out_size, void* d_ws, size_t ws_size,
                              hipStream_t stream) {
  (void)in_sizes; (void)n_in; (void)out_size; (void)ws_size;
  const float* xyz = (const float*)d_in[0];
  const float* W1  = (const float*)d_in[1];
  const float* b1  = (const float*)d_in[2];
  const float* g1  = (const float*)d_in[3];
  const float* be1 = (const float*)d_in[4];
  const float* W2  = (const float*)d_in[5];
  const float* b2  = (const float*)d_in[6];
  const float* W3  = (const float*)d_in[7];
  const float* b3  = (const float*)d_in[8];
  const float* g3  = (const float*)d_in[9];
  const float* be3 = (const float*)d_in[10];
  const float* W4  = (const float*)d_in[11];
  const float* b4  = (const float*)d_in[12];

  const size_t M = MROWS;
  char* ws = (char*)d_ws;
  size_t off = 0;
  auto alloc = [&](size_t bytes) { size_t o = off; off += (bytes + 255) & ~(size_t)255; return o; };

  int*    idxw   = (int*)   (ws + alloc(M * 4));
  bf16_t* h1     = (bf16_t*)(ws + alloc(M * 128 * 2));      // BN1 applied in place
  bf16_t* h2     = (bf16_t*)(ws + alloc(M * 256 * 2));
  bf16_t* gmaxb  = (bf16_t*)(ws + alloc((size_t)PPTS * 256 * 2));
  bf16_t* f3     = (bf16_t*)(ws + alloc(M * 512 * 2));      // BN3 applied in place
  bf16_t* w2t    = (bf16_t*)(ws + alloc(256 * 128 * 2));
  bf16_t* w3t    = (bf16_t*)(ws + alloc(512 * 512 * 2));
  bf16_t* w4t    = (bf16_t*)(ws + alloc((size_t)HDIM * 512 * 2));
  float*  stats1 = (float*) (ws + alloc(256 * 4));
  float*  stats3 = (float*) (ws + alloc(1024 * 4));
  float*  sc1    = (float*) (ws + alloc(128 * 4));
  float*  sh1    = (float*) (ws + alloc(128 * 4));
  float*  sc3    = (float*) (ws + alloc(512 * 4));
  float*  sh3    = (float*) (ws + alloc(512 * 4));

  float*    out_xyz = (float*)d_out;
  unsigned* enc     = (unsigned*)d_out + (size_t)BATCH * NPTS * 3;
  const size_t feat_total = (size_t)PPTS * HDIM;   // 3,145,728
  const float invM = 1.0f / (float)M;

  init_kernel<<<dim3((unsigned)((feat_total + 255) / 256)), 256, 0, stream>>>(
      xyz, out_xyz, enc, stats1, stats3);

  transpose_bf16_kernel<<<dim3((128 * 256 + 255) / 256), 256, 0, stream>>>(W2, w2t, 128, 256);
  transpose_bf16_kernel<<<dim3((512 * 512 + 255) / 256), 256, 0, stream>>>(W3, w3t, 512, 512);
  transpose_bf16_kernel<<<dim3((512 * HDIM + 255) / 256), 256, 0, stream>>>(W4, w4t, 512, HDIM);

  knn_kernel<<<dim3(NPTS / 256, BATCH), 256, 0, stream>>>(xyz, idxw);

  mlp1_kernel<<<dim3((unsigned)(M / 128)), 128, 0, stream>>>(xyz, idxw, W1, b1, h1, stats1);
  bn_finalize_kernel<<<1, 128, 0, stream>>>(stats1, 128, invM, g1, be1, sc1, sh1);
  bn_apply_kernel<<<dim3(2048), 256, 0, stream>>>(h1, sc1, sh1, 127u, M * 128);

  gemm2_kernel<<<dim3(320, 4), 128, 0, stream>>>(h1, w2t, b2, h2);
  gmax_kernel<<<dim3(PPTS), 256, 0, stream>>>(h2, gmaxb);

  gemm3_kernel<<<dim3(320, 8), 128, 0, stream>>>(gmaxb, h2, w3t, b3, f3, stats3);
  bn_finalize_kernel<<<1, 512, 0, stream>>>(stats3, 512, invM, g3, be3, sc3, sh3);
  bn_apply_kernel<<<dim3(4096), 256, 0, stream>>>(f3, sc3, sh3, 511u, M * 512);

  gemm4_kernel<<<dim3(320, 6), 128, 0, stream>>>(f3, w4t, b4, enc);
  decode_kernel<<<dim3((unsigned)((feat_total + 255) / 256)), 256, 0, stream>>>(enc, feat_total);
}
